// Transformer_23502061043767
// MI455X (gfx1250) — matrix operand order repaired and checked by reading, not compiled, not run on hardware
//
#include <hip/hip_runtime.h>
#include <hip/hip_bf16.h>
#include <math.h>

typedef __attribute__((ext_vector_type(16))) __bf16 bf16x16;
typedef __attribute__((ext_vector_type(8)))  __bf16 bf16x8;
typedef __attribute__((ext_vector_type(8)))  float  f32x8;
typedef __attribute__((ext_vector_type(4)))  unsigned int u32x4;
typedef __attribute__((ext_vector_type(4)))  int v4i;
typedef unsigned short ushort_t;

__device__ __forceinline__ ushort_t f2bf_bits(float f) {
    unsigned u = __float_as_uint(f);
    unsigned rnd = 0x7fffu + ((u >> 16) & 1u);   // round-to-nearest-even
    return (ushort_t)((u + rnd) >> 16);
}
__device__ __forceinline__ float bf2f(ushort_t b) {
    return __uint_as_float(((unsigned)b) << 16);
}

// ---- CDNA5 async global->LDS copy (DMA path, ASYNCcnt), with sync fallback ----
#if defined(__has_builtin)
#if __has_builtin(__builtin_amdgcn_global_load_async_to_lds_b128)
#define HAS_ASYNC_LDS 1
#endif
#endif

#ifdef HAS_ASYNC_LDS
__device__ __forceinline__ void cp16_g2l(const void* g, void* l) {
    __builtin_amdgcn_global_load_async_to_lds_b128(
        (__attribute__((address_space(1))) v4i*)(g),
        (__attribute__((address_space(3))) v4i*)(l), 0, 0);
}
__device__ __forceinline__ void wait_async() {
    asm volatile("s_wait_asynccnt 0x0" ::: "memory");
}
#else
__device__ __forceinline__ void cp16_g2l(const void* g, void* l) {
    *(u32x4*)l = *(const u32x4*)g;
}
__device__ __forceinline__ void wait_async() {}
#endif

// ---------------------------------------------------------------------------
// f32 -> bf16 conversion (grid-stride, memory bound)
// ---------------------------------------------------------------------------
__global__ __launch_bounds__(256) void cvt_f32_bf16(const float* __restrict__ in,
                                                    ushort_t* __restrict__ out, int n) {
    for (int i = blockIdx.x * 256 + threadIdx.x; i < n; i += gridDim.x * 256)
        out[i] = f2bf_bits(in[i]);
}

// ---------------------------------------------------------------------------
// Generic NT GEMM: C[M,N] = alpha * (A[M,K] @ B[N,K]^T) + bias + Cadd, epilogue
// A,B bf16 row-major (K contiguous). 128x128 block tile, 8 waves of 64x32,
// v_wmma_f32_16x16x32_bf16 with f32 accumulate. Double-buffered LDS tiles
// filled by async global->LDS DMA (ASYNCcnt), one barrier per K-tile.
// op: 0=none, 1=sigmoid, 2=relu. Outputs: Cf (f32), Cb (bf16), CbT (bf16 transposed)
// All dims are multiples of the tile sizes for this problem.
// ---------------------------------------------------------------------------
#define TK 64
#define LDP (TK + 8)   // padded LDS row stride (elements); 144B, 16B-aligned
__global__ __launch_bounds__(256, 2)
void gemm_bf16_nt(const ushort_t* __restrict__ A, const ushort_t* __restrict__ B,
                  float* __restrict__ Cf, ushort_t* __restrict__ Cb, ushort_t* __restrict__ CbT,
                  const float* __restrict__ bias, const float* __restrict__ Cadd,
                  int M, int N, int K, float alpha, int op) {
    __shared__ ushort_t Alds[2][128][LDP];
    __shared__ ushort_t Blds[2][128][LDP];

    const int tid  = threadIdx.x;
    const int lane = tid & 31;
    const int wave = tid >> 5;      // 0..7
    const int wm   = wave & 1;      // 2 wave-rows  -> 64 M each
    const int wn   = wave >> 1;     // 4 wave-cols  -> 32 N each
    const int bm   = blockIdx.y * 128;
    const int bn   = blockIdx.x * 128;

    f32x8 acc[4][2];
#pragma unroll
    for (int i = 0; i < 4; ++i)
#pragma unroll
        for (int j = 0; j < 2; ++j)
#pragma unroll
            for (int v = 0; v < 8; ++v) acc[i][j][v] = 0.0f;

    // Tile staging: 128 rows x 64 elems x 2B = 16KB per matrix = 1024 16B-chunks,
    // 256 threads -> 4 chunks each per matrix.
    auto stage = [&](int buf, int k0) {
#pragma unroll
        for (int p = 0; p < 4; ++p) {
            const int chunk = tid + p * 256;       // 0..1023
            const int row   = chunk >> 3;          // 0..127
            const int col   = (chunk & 7) * 8;     // 0..56 step 8
            cp16_g2l(&A[(size_t)(bm + row) * K + k0 + col], &Alds[buf][row][col]);
            cp16_g2l(&B[(size_t)(bn + row) * K + k0 + col], &Blds[buf][row][col]);
        }
    };

    stage(0, 0);
    int buf = 0;
    for (int k0 = 0; k0 < K; k0 += TK) {
        wait_async();
        __syncthreads();
        if (k0 + TK < K) stage(buf ^ 1, k0 + TK);   // async fill of next tile

#pragma unroll
        for (int s = 0; s < 2; ++s) {               // two K=32 WMMA steps
            const int kbase = s * 32;
            // B fragments: lane<16 holds K=0..15 of col (lane&15); lane>=16 K=16..31
            bf16x16 bfrag[2];
#pragma unroll
            for (int jn = 0; jn < 2; ++jn) {
                const int ncol = wn * 32 + jn * 16 + (lane & 15);
                const int kb   = kbase + (lane >> 4) * 16;
                bf16x8 blo = *(const bf16x8*)&Blds[buf][ncol][kb];
                bf16x8 bhi = *(const bf16x8*)&Blds[buf][ncol][kb + 8];
#pragma unroll
                for (int q = 0; q < 8; ++q) { bfrag[jn][q] = blo[q]; bfrag[jn][8 + q] = bhi[q]; }
            }
            // A fragments: lane<16 holds K=0..15 of row (lane&15); lane>=16 K=16..31
#pragma unroll
            for (int i = 0; i < 4; ++i) {
                const int mrow = wm * 64 + i * 16 + (lane & 15);
                const int ka   = kbase + (lane >> 4) * 16;
                bf16x8 alo = *(const bf16x8*)&Alds[buf][mrow][ka];
                bf16x8 ahi = *(const bf16x8*)&Alds[buf][mrow][ka + 8];
                bf16x16 afrag;
#pragma unroll
                for (int q = 0; q < 8; ++q) { afrag[q] = alo[q]; afrag[8 + q] = ahi[q]; }
#pragma unroll
                for (int jn = 0; jn < 2; ++jn)
                    acc[i][jn] = __builtin_amdgcn_wmma_f32_16x16x32_bf16(
                        false, afrag, false, bfrag[jn], (short)0, acc[i][jn], false, false);
            }
        }
        buf ^= 1;
    }

    // Epilogue. C layout: acc elem v -> row = base + (lane>>4)*8 + v, col = base + (lane&15)
    const int r0 = bm + wm * 64;
    const int c0 = bn + wn * 32;
#pragma unroll
    for (int i = 0; i < 4; ++i) {
#pragma unroll
        for (int jn = 0; jn < 2; ++jn) {
            const int colg = c0 + jn * 16 + (lane & 15);
            const float bval = bias ? bias[colg] : 0.0f;
#pragma unroll
            for (int v = 0; v < 8; ++v) {
                const int rowg = r0 + i * 16 + ((lane >> 4) * 8) + v;
                float x = acc[i][jn][v] * alpha + bval;
                if (Cadd) x += Cadd[(size_t)rowg * N + colg];
                if (op == 1)      x = 1.0f / (1.0f + __expf(-x));
                else if (op == 2) x = fmaxf(x, 0.0f);
                if (Cf)  Cf [(size_t)rowg * N + colg] = x;
                if (Cb)  Cb [(size_t)rowg * N + colg] = f2bf_bits(x);
                if (CbT) CbT[(size_t)colg * M + rowg] = f2bf_bits(x);  // transposed store
            }
        }
    }
}

// ---------------------------------------------------------------------------
// Row softmax fused with gate multiply: G <- bf16( G * softmax(S_row) )
// One block per row; deterministic tree reductions; exp values cached in LDS.
// ---------------------------------------------------------------------------
__global__ __launch_bounds__(256) void softmax_gate_kernel(const float* __restrict__ S,
                                                           ushort_t* __restrict__ G, int N) {
    __shared__ float buf[4096];
    __shared__ float red[256];
    const int tid = threadIdx.x;
    const int r   = blockIdx.x;
    const float* srow = S + (size_t)r * N;

    float mx = -3.4e38f;
    for (int c = tid; c < N; c += 256) mx = fmaxf(mx, srow[c]);
    red[tid] = mx; __syncthreads();
    for (int s = 128; s > 0; s >>= 1) { if (tid < s) red[tid] = fmaxf(red[tid], red[tid + s]); __syncthreads(); }
    const float rowmax = red[0];
    __syncthreads();

    float sum = 0.0f;
    for (int c = tid; c < N; c += 256) { float ev = __expf(srow[c] - rowmax); buf[c] = ev; sum += ev; }
    red[tid] = sum; __syncthreads();
    for (int s = 128; s > 0; s >>= 1) { if (tid < s) red[tid] += red[tid + s]; __syncthreads(); }
    const float inv = 1.0f / red[0];

    ushort_t* grow = G + (size_t)r * N;
    for (int c = tid; c < N; c += 256) {
        float g = bf2f(grow[c]);
        grow[c] = f2bf_bits(g * buf[c] * inv);
    }
}

// ---------------------------------------------------------------------------
// BatchNorm over rows (axis 0), biased variance. Deterministic.
// stats: one block handles 32 columns; 8 row-lanes x 32 col-lanes (coalesced).
// ---------------------------------------------------------------------------
__global__ __launch_bounds__(256) void bn_stats(const float* __restrict__ X,
                                                float* __restrict__ mu, float* __restrict__ rstd,
                                                int Nrows, int D) {
    __shared__ float sb[8][32], qb[8][32];
    const int cx = threadIdx.x & 31;
    const int ty = threadIdx.x >> 5;
    const int c  = blockIdx.x * 32 + cx;
    float s = 0.0f, q = 0.0f;
    for (int r = ty; r < Nrows; r += 8) {
        float v = X[(size_t)r * D + c];
        s += v; q += v * v;
    }
    sb[ty][cx] = s; qb[ty][cx] = q; __syncthreads();
    if (ty == 0) {
        for (int k = 1; k < 8; ++k) { s += sb[k][cx]; q += qb[k][cx]; }
        float m   = s / (float)Nrows;
        float var = q / (float)Nrows - m * m;
        mu[c]   = m;
        rstd[c] = rsqrtf(var + 1e-5f);
    }
}

__global__ __launch_bounds__(256) void bn_apply(const float* __restrict__ X,
                                                const float* __restrict__ mu, const float* __restrict__ rstd,
                                                const float* __restrict__ gamma, const float* __restrict__ beta,
                                                float* __restrict__ Yf, ushort_t* __restrict__ Yb,
                                                int Nrows, int D) {
    const int total = Nrows * D;
    for (int i = blockIdx.x * 256 + threadIdx.x; i < total; i += gridDim.x * 256) {
        const int c = i % D;
        float v = (X[i] - mu[c]) * rstd[c] * gamma[c] + beta[c];
        if (Yf) Yf[i] = v;
        if (Yb) Yb[i] = f2bf_bits(v);
    }
}

// ---------------------------------------------------------------------------
// Host-side orchestration
// ---------------------------------------------------------------------------
extern "C" void kernel_launch(void* const* d_in, const int* in_sizes, int n_in,
                              void* d_out, int out_size, void* d_ws, size_t ws_size,
                              hipStream_t stream) {
    (void)in_sizes; (void)n_in; (void)out_size; (void)ws_size;
    const int N = 4096, D = 512, E = 4096, D2 = 1024;

    const float* h     = (const float*)d_in[0];
    const float* e     = (const float*)d_in[1];
    const float* Wq    = (const float*)d_in[2];
    const float* bq    = (const float*)d_in[3];
    const float* Wk    = (const float*)d_in[4];
    const float* bk    = (const float*)d_in[5];
    const float* Wv    = (const float*)d_in[6];
    const float* bv    = (const float*)d_in[7];
    const float* Wpe   = (const float*)d_in[8];
    const float* bpe   = (const float*)d_in[9];
    const float* Wge   = (const float*)d_in[10];
    const float* bge   = (const float*)d_in[11];
    const float* W0    = (const float*)d_in[12];
    const float* b0    = (const float*)d_in[13];
    const float* W1    = (const float*)d_in[14];
    const float* b1    = (const float*)d_in[15];
    const float* W2    = (const float*)d_in[16];
    const float* b2    = (const float*)d_in[17];
    const float* gamma = (const float*)d_in[18];
    const float* beta  = (const float*)d_in[19];
    float* out = (float*)d_out;

    char* ws = (char*)d_ws;
    const size_t MB = 1ull << 20;
    // Long-lived region
    ushort_t* e_bf   = (ushort_t*)(ws + 0);            // 32 MB
    ushort_t* Wbig   = (ushort_t*)(ws + 32 * MB);      // 32 MB (Wpe then Wge)
    float*    S      = (float*)   (ws + 64 * MB);      // 64 MB scores/proj_e (freed after softmax)
    ushort_t* Gp     = (ushort_t*)(ws + 128 * MB);     // 32 MB gate -> P (in place)
    ushort_t* h_bf   = (ushort_t*)(ws + 160 * MB);     // 4 MB
    ushort_t* Wq_bf  = (ushort_t*)(ws + 164 * MB);
    ushort_t* Wk_bf  = (ushort_t*)(ws + 164 * MB + 512 * 1024);
    ushort_t* Wv_bf  = (ushort_t*)(ws + 165 * MB);
    ushort_t* W0_bf  = (ushort_t*)(ws + 165 * MB + 512 * 1024);
    ushort_t* W1_bf  = (ushort_t*)(ws + 166 * MB);     // 1 MB
    ushort_t* W2_bf  = (ushort_t*)(ws + 167 * MB);     // 1 MB
    ushort_t* Qh_bf  = (ushort_t*)(ws + 168 * MB);     // 4 MB
    ushort_t* Kh_bf  = (ushort_t*)(ws + 172 * MB);     // 4 MB
    ushort_t* VhT_bf = (ushort_t*)(ws + 176 * MB);     // 4 MB  (transposed: [D][N])
    float*    mu     = (float*)   (ws + 180 * MB);
    float*    rstd   = (float*)   (ws + 180 * MB + 4096);
    // Region reusing S after softmax consumed it
    float*    attn_f  = (float*)   (ws + 64 * MB);     // 8 MB
    ushort_t* attn_bf = (ushort_t*)(ws + 72 * MB);     // 4 MB
    float*    x_f     = (float*)   (ws + 76 * MB);     // 8 MB
    float*    res_f   = (float*)   (ws + 84 * MB);     // 8 MB
    ushort_t* xbn_bf  = (ushort_t*)(ws + 92 * MB);     // 4 MB
    ushort_t* y1_bf   = (ushort_t*)(ws + 96 * MB);     // 8 MB
    float*    x2_f    = (float*)   (ws + 104 * MB);    // 8 MB

    const float inv_sqrt_d = 0.044194173824159216f;    // 1/sqrt(512)
    dim3 blk(256);
    auto cvt = [&](const float* src, ushort_t* dst, int n) {
        cvt_f32_bf16<<<dim3(4096), blk, 0, stream>>>(src, dst, n);
    };

    // --- convert operands to bf16 ---
    cvt(h, h_bf, N * D);
    cvt(Wq, Wq_bf, D * D);  cvt(Wk, Wk_bf, D * D);  cvt(Wv, Wv_bf, D * D);
    cvt(W0, W0_bf, D * D);  cvt(W1, W1_bf, D2 * D); cvt(W2, W2_bf, D * D2);
    cvt(e, e_bf, N * E);
    cvt(Wpe, Wbig, E * E);

    // --- proj_e = e @ Wpe^T + bpe  -> S (f32) ---
    gemm_bf16_nt<<<dim3(E / 128, N / 128), blk, 0, stream>>>(
        e_bf, Wbig, S, nullptr, nullptr, bpe, nullptr, N, E, E, 1.0f, 0);

    // --- gate = sigmoid(e @ Wge^T + bge) -> Gp (bf16) ---
    cvt(Wge, Wbig, E * E);
    gemm_bf16_nt<<<dim3(E / 128, N / 128), blk, 0, stream>>>(
        e_bf, Wbig, nullptr, Gp, nullptr, bge, nullptr, N, E, E, 1.0f, 1);

    // --- Qh, Kh (bf16), Vh (bf16 transposed) ---
    gemm_bf16_nt<<<dim3(D / 128, N / 128), blk, 0, stream>>>(
        h_bf, Wq_bf, nullptr, Qh_bf, nullptr, bq, nullptr, N, D, D, 1.0f, 0);
    gemm_bf16_nt<<<dim3(D / 128, N / 128), blk, 0, stream>>>(
        h_bf, Wk_bf, nullptr, Kh_bf, nullptr, bk, nullptr, N, D, D, 1.0f, 0);
    gemm_bf16_nt<<<dim3(D / 128, N / 128), blk, 0, stream>>>(
        h_bf, Wv_bf, nullptr, nullptr, VhT_bf, bv, nullptr, N, D, D, 1.0f, 0);

    // --- scores: S <- (Qh @ Kh^T)/sqrt(d) + S (in place) ---
    gemm_bf16_nt<<<dim3(N / 128, N / 128), blk, 0, stream>>>(
        Qh_bf, Kh_bf, S, nullptr, nullptr, nullptr, S, N, N, D, inv_sqrt_d, 0);

    // --- P = gate * softmax(S) (bf16, overwrites Gp) ---
    softmax_gate_kernel<<<dim3(N), blk, 0, stream>>>(S, Gp, N);

    // --- attn_out = P @ Vh  (NT vs VhT) -> f32 + bf16 ---
    gemm_bf16_nt<<<dim3(D / 128, N / 128), blk, 0, stream>>>(
        Gp, VhT_bf, attn_f, attn_bf, nullptr, nullptr, nullptr, N, D, N, 1.0f, 0);

    // --- x = attn_out + attn_out @ W0^T + b0 ---
    gemm_bf16_nt<<<dim3(D / 128, N / 128), blk, 0, stream>>>(
        attn_bf, W0_bf, x_f, nullptr, nullptr, b0, attn_f, N, D, D, 1.0f, 0);

    // --- BatchNorm #1 -> res (f32) + bf16 copy ---
    bn_stats<<<dim3(D / 32), blk, 0, stream>>>(x_f, mu, rstd, N, D);
    bn_apply<<<dim3(2048), blk, 0, stream>>>(x_f, mu, rstd, gamma, beta, res_f, xbn_bf, N, D);

    // --- y1 = relu(res @ W1^T + b1) (bf16) ---
    gemm_bf16_nt<<<dim3(D2 / 128, N / 128), blk, 0, stream>>>(
        xbn_bf, W1_bf, nullptr, y1_bf, nullptr, b1, nullptr, N, D2, D, 1.0f, 2);

    // --- x2 = y1 @ W2^T + b2 + res ---
    gemm_bf16_nt<<<dim3(D / 128, N / 128), blk, 0, stream>>>(
        y1_bf, W2_bf, x2_f, nullptr, nullptr, b2, res_f, N, D, D2, 1.0f, 0);

    // --- BatchNorm #2 -> d_out (f32) ---
    bn_stats<<<dim3(D / 32), blk, 0, stream>>>(x2_f, mu, rstd, N, D);
    bn_apply<<<dim3(2048), blk, 0, stream>>>(x2_f, mu, rstd, gamma, beta, out, nullptr, N, D);
}
